// MHSA_59390807769232
// MI455X (gfx1250) — matrix operand order repaired and checked by reading, not compiled, not run on hardware
//
#include <hip/hip_runtime.h>
#include <hip/hip_bf16.h>
#include <math.h>

// ---------------------------------------------------------------- constants
#define BATCH 4
#define SEQ   2048
#define DM    1024
#define NH    16
#define DK    64
#define QKVN  3072
#define ROWS  (BATCH * SEQ)          // 8192

typedef __bf16 v16bf __attribute__((ext_vector_type(16)));
typedef __bf16 bf16x8 __attribute__((ext_vector_type(8)));
typedef float  v8f   __attribute__((ext_vector_type(8)));

#define V8F_ZERO {0.f,0.f,0.f,0.f,0.f,0.f,0.f,0.f}

// ------------------------------------------------------------- WMMA helpers
__device__ __forceinline__ v8f wmma_bf16(v16bf a, v16bf b, v8f c) {
  // D = A(16x32,bf16) * B(32x16,bf16) + C(16x16,f32)
  return __builtin_amdgcn_wmma_f32_16x16x32_bf16(
      /*neg_a=*/false, a, /*neg_b=*/false, b,
      /*c_mod=*/(short)0, c, /*reuse_a=*/false, /*reuse_b=*/false);
}

// A-matrix 16x32 bf16 fragment (ISA 7.12.2): lane l holds row (l&15),
// K chunks {c0..c0+7, c0+16..c0+23} with c0 = (l>>4)*8.
__device__ __forceinline__ v16bf load_fragA(const __bf16* base, int stride, int lane) {
  const int r  = lane & 15;
  const int c0 = (lane >> 4) * 8;
  bf16x8 lo = *(const bf16x8*)(base + (size_t)r * stride + c0);
  bf16x8 hi = *(const bf16x8*)(base + (size_t)r * stride + c0 + 16);
  return __builtin_shufflevector(lo, hi, 0,1,2,3,4,5,6,7,8,9,10,11,12,13,14,15);
}

// B-matrix 32x16 bf16 fragment: lane l holds column (l&15) with
// K chunks {k0..k0+7, k0+16..k0+23}, k0 = (l>>4)*8.  base must point at column-major data (row n of the
// transposed operand), stride in elements, 16B aligned rows.
__device__ __forceinline__ v16bf load_fragB(const __bf16* base, int stride, int lane) {
  const int n  = lane & 15;
  const int k0 = (lane >> 4) * 8;
  bf16x8 lo = *(const bf16x8*)(base + (size_t)n * stride + k0);
  bf16x8 hi = *(const bf16x8*)(base + (size_t)n * stride + k0 + 16);
  return __builtin_shufflevector(lo, hi, 0,1,2,3,4,5,6,7,8,9,10,11,12,13,14,15);
}

__device__ __forceinline__ float rowmax16(float v) {
  v = fmaxf(v, __shfl_xor(v, 1, 32));
  v = fmaxf(v, __shfl_xor(v, 2, 32));
  v = fmaxf(v, __shfl_xor(v, 4, 32));
  v = fmaxf(v, __shfl_xor(v, 8, 32));
  return v;
}
__device__ __forceinline__ float rowsum16(float v) {
  v += __shfl_xor(v, 1, 32);
  v += __shfl_xor(v, 2, 32);
  v += __shfl_xor(v, 4, 32);
  v += __shfl_xor(v, 8, 32);
  return v;
}

// ------------------------------------------------------------ convert f32->bf16
__global__ void f32_to_bf16_kernel(const float* __restrict__ in,
                                   __bf16* __restrict__ out, int n) {
  int i = blockIdx.x * blockDim.x + threadIdx.x;
  int stride = gridDim.x * blockDim.x;
  for (; i < n; i += stride) out[i] = (__bf16)in[i];
}

// ------------------------------------------------------------ GEMM: C = A * B^T
// A: MxK bf16 row-major, B: NxK bf16 row-major (so B^T columns = B rows),
// C: MxN (OutT). Block tile 64x128, 8 waves of 32x32, k-step 32.
template <typename OutT>
__global__ __launch_bounds__(256)
void gemm_bt_kernel(const __bf16* __restrict__ A, const __bf16* __restrict__ B,
                    OutT* __restrict__ C, int M, int N, int K) {
  __shared__ __attribute__((aligned(32))) __bf16 As[64][40];   // 16B-aligned rows
  __shared__ __attribute__((aligned(32))) __bf16 Bs[128][48];  // 32B-aligned rows

  const int t    = threadIdx.x;
  const int lane = t & 31;
  const int wave = t >> 5;
  const int wm   = wave >> 2;               // 0..1
  const int wn   = wave & 3;                // 0..3
  const int m0   = blockIdx.y * 64;
  const int n0   = blockIdx.x * 128;

  v8f acc00 = V8F_ZERO, acc01 = V8F_ZERO, acc10 = V8F_ZERO, acc11 = V8F_ZERO;

  const int arow = t >> 2;                  // 0..63
  const int acol = (t & 3) * 8;

  for (int k0 = 0; k0 < K; k0 += 32) {
    __syncthreads();
    // stage A tile (64x32)
    *(bf16x8*)&As[arow][acol] =
        *(const bf16x8*)&A[(size_t)(m0 + arow) * K + k0 + acol];
    // stage B tile (128x32) : 2 chunks per thread
    {
      int id = t;
      int r = id >> 2, c = (id & 3) * 8;
      *(bf16x8*)&Bs[r][c] = *(const bf16x8*)&B[(size_t)(n0 + r) * K + k0 + c];
      id = t + 256; r = id >> 2; c = (id & 3) * 8;
      *(bf16x8*)&Bs[r][c] = *(const bf16x8*)&B[(size_t)(n0 + r) * K + k0 + c];
    }
    __syncthreads();

    v16bf a0 = load_fragA(&As[wm * 32     ][0], 40, lane);
    v16bf a1 = load_fragA(&As[wm * 32 + 16][0], 40, lane);
    v16bf b0 = load_fragB(&Bs[wn * 32     ][0], 48, lane);
    v16bf b1 = load_fragB(&Bs[wn * 32 + 16][0], 48, lane);

    acc00 = wmma_bf16(a0, b0, acc00);
    acc01 = wmma_bf16(a0, b1, acc01);
    acc10 = wmma_bf16(a1, b0, acc10);
    acc11 = wmma_bf16(a1, b1, acc11);
  }

  // epilogue: C layout: vgpr i, lanes 0-15 -> M=i, lanes 16-31 -> M=i+8
  const int half = lane >> 4, nl = lane & 15;
  v8f accs[2][2] = {{acc00, acc01}, {acc10, acc11}};
#pragma unroll
  for (int tm = 0; tm < 2; ++tm)
#pragma unroll
    for (int tn = 0; tn < 2; ++tn) {
      const int gm0 = m0 + wm * 32 + tm * 16 + 8 * half;
      const int gn0 = n0 + wn * 32 + tn * 16 + nl;
#pragma unroll
      for (int i = 0; i < 8; ++i)
        C[(size_t)(gm0 + i) * N + gn0] = (OutT)accs[tm][tn][i];
    }
}

// ------------------------------------------------------------ RoPE + repack
// qkv: [ROWS][3072] bf16 -> Qr/Kr: [b][h][s][dk] bf16 (roped),
//                           Vt:   [b][h][dk][s] bf16 (transposed)
__global__ void rope_pack_kernel(const __bf16* __restrict__ qkv,
                                 const int* __restrict__ tpos,
                                 __bf16* __restrict__ Qr,
                                 __bf16* __restrict__ Kr,
                                 __bf16* __restrict__ Vt) {
  int idx = blockIdx.x * blockDim.x + threadIdx.x;   // B*S*H*32 = 2^22
  if (idx >= BATCH * SEQ * NH * 32) return;
  const int p = idx & 31;
  int tmp = idx >> 5;
  const int h = tmp & (NH - 1);  tmp >>= 4;
  const int s = tmp & (SEQ - 1);
  const int b = tmp >> 11;

  const __bf16* row = qkv + (size_t)(b * SEQ + s) * QKVN;
  const float q1 = (float)row[h * DK + 2 * p];
  const float q2 = (float)row[h * DK + 2 * p + 1];
  const float k1 = (float)row[DM + h * DK + 2 * p];
  const float k2 = (float)row[DM + h * DK + 2 * p + 1];
  const float v1 = (float)row[2 * DM + h * DK + 2 * p];
  const float v2 = (float)row[2 * DM + h * DK + 2 * p + 1];

  // freq = THETA^(-2p/dk) = exp(-p * ln(10000)/32)
  const float freq = __expf(-0.28782313662425572f * (float)p);
  const float ang  = (float)tpos[s] * freq;
  float sn, cs;
  __sincosf(ang, &sn, &cs);

  const size_t qkbase = ((size_t)(b * NH + h) * SEQ + s) * DK;
  Qr[qkbase + 2 * p]     = (__bf16)(q1 * cs - q2 * sn);
  Qr[qkbase + 2 * p + 1] = (__bf16)(q1 * sn + q2 * cs);
  Kr[qkbase + 2 * p]     = (__bf16)(k1 * cs - k2 * sn);
  Kr[qkbase + 2 * p + 1] = (__bf16)(k1 * sn + k2 * cs);

  const size_t vtbase = (size_t)(b * NH + h) * DK * SEQ;
  Vt[vtbase + (size_t)(2 * p)     * SEQ + s] = (__bf16)v1;
  Vt[vtbase + (size_t)(2 * p + 1) * SEQ + s] = (__bf16)v2;
}

// ------------------------------------------------------------ flash attention
// one wave per (b, h, 16-query tile); causal; online softmax; WMMA for
// scores and P*V.  Output: [b*S][DM] bf16 (heads re-interleaved).
__global__ __launch_bounds__(128)
void attn_kernel(const __bf16* __restrict__ Qr, const __bf16* __restrict__ Kr,
                 const __bf16* __restrict__ Vt, __bf16* __restrict__ Ob) {
  __shared__ __attribute__((aligned(32))) __bf16 Pst[4][16][40];

  const int wave = threadIdx.x >> 5;
  const int lane = threadIdx.x & 31;
  const int job  = blockIdx.x * 4 + wave;          // 8192 jobs
  const int mt = job & 127;
  const int h  = (job >> 7) & (NH - 1);
  const int b  = job >> 11;

  const __bf16* Q = Qr + (size_t)(b * NH + h) * SEQ * DK;
  const __bf16* K = Kr + (size_t)(b * NH + h) * SEQ * DK;
  const __bf16* V = Vt + (size_t)(b * NH + h) * DK * SEQ;

  // Q fragments: 16 rows x 64 dk -> two 16x32 A-frags
  const v16bf qa0 = load_fragA(Q + (size_t)(mt * 16) * DK +  0, DK, lane);
  const v16bf qa1 = load_fragA(Q + (size_t)(mt * 16) * DK + 32, DK, lane);

  v8f o0 = V8F_ZERO, o1 = V8F_ZERO, o2 = V8F_ZERO, o3 = V8F_ZERO;
  float mrow[8], lrow[8];
#pragma unroll
  for (int i = 0; i < 8; ++i) { mrow[i] = -1e30f; lrow[i] = 0.f; }

  const int half = lane >> 4, nl = lane & 15;
  const int nblocks = (mt >> 1) + 1;               // causal: keys <= mt*16+15

  for (int blk = 0; blk < nblocks; ++blk) {
    const int j = blk * 32;

    // ---- scores: S(16x32) = Q(16x64) @ K^T(64x32), two 16-key n-tiles
    v8f s0 = V8F_ZERO, s1 = V8F_ZERO;
    {
      v16bf kb0a = load_fragB(K + (size_t)j        * DK +  0, DK, lane);
      v16bf kb0b = load_fragB(K + (size_t)j        * DK + 32, DK, lane);
      v16bf kb1a = load_fragB(K + (size_t)(j + 16) * DK +  0, DK, lane);
      v16bf kb1b = load_fragB(K + (size_t)(j + 16) * DK + 32, DK, lane);
      s0 = wmma_bf16(qa0, kb0a, s0);
      s0 = wmma_bf16(qa1, kb0b, s0);
      s1 = wmma_bf16(qa0, kb1a, s1);
      s1 = wmma_bf16(qa1, kb1b, s1);
    }

    // ---- scale + causal mask + online softmax update
#pragma unroll
    for (int i = 0; i < 8; ++i) {
      const int qi = mt * 16 + i + 8 * half;       // global query index
      float v0 = s0[i] * 0.125f;                   // 1/sqrt(64)
      float v1 = s1[i] * 0.125f;
      if (j + nl      > qi) v0 = -1e30f;
      if (j + 16 + nl > qi) v1 = -1e30f;

      float bm   = rowmax16(fmaxf(v0, v1));
      float mnew = fmaxf(mrow[i], bm);
      float alpha = (mrow[i] > -5e29f) ? __expf(mrow[i] - mnew) : 0.f;
      float p0 = (v0 > -5e29f) ? __expf(v0 - mnew) : 0.f;
      float p1 = (v1 > -5e29f) ? __expf(v1 - mnew) : 0.f;
      float rs = rowsum16(p0 + p1);

      mrow[i] = mnew;
      lrow[i] = lrow[i] * alpha + rs;
      o0[i] *= alpha; o1[i] *= alpha; o2[i] *= alpha; o3[i] *= alpha;

      // stage P (C-layout -> row-major LDS) for A-fragment reload
      Pst[wave][i + 8 * half][nl]      = (__bf16)p0;
      Pst[wave][i + 8 * half][16 + nl] = (__bf16)p1;
    }

    asm volatile("s_wait_dscnt 0" ::: "memory");   // LDS stores -> loads

    // ---- O += P(16x32) @ V(32x64): 4 n-tiles over dk
    v16bf pf  = load_fragA(&Pst[wave][0][0], 40, lane);
    v16bf vb0 = load_fragB(V + (size_t) 0 * SEQ + j, SEQ, lane);
    v16bf vb1 = load_fragB(V + (size_t)16 * SEQ + j, SEQ, lane);
    v16bf vb2 = load_fragB(V + (size_t)32 * SEQ + j, SEQ, lane);
    v16bf vb3 = load_fragB(V + (size_t)48 * SEQ + j, SEQ, lane);
    o0 = wmma_bf16(pf, vb0, o0);
    o1 = wmma_bf16(pf, vb1, o1);
    o2 = wmma_bf16(pf, vb2, o2);
    o3 = wmma_bf16(pf, vb3, o3);
  }

  // ---- normalize and write back as [b*S][DM] bf16
#pragma unroll
  for (int i = 0; i < 8; ++i) {
    const float inv = 1.f / lrow[i];
    const size_t orow = (size_t)(b * SEQ + mt * 16 + i + 8 * half) * DM
                      + (size_t)h * DK + nl;
    Ob[orow +  0] = (__bf16)(o0[i] * inv);
    Ob[orow + 16] = (__bf16)(o1[i] * inv);
    Ob[orow + 32] = (__bf16)(o2[i] * inv);
    Ob[orow + 48] = (__bf16)(o3[i] * inv);
  }
}

// ------------------------------------------------------------ launch
extern "C" void kernel_launch(void* const* d_in, const int* in_sizes, int n_in,
                              void* d_out, int out_size, void* d_ws, size_t ws_size,
                              hipStream_t stream) {
  const float* x     = (const float*)d_in[0];    // [4,2048,1024]
  const float* Wqkv  = (const float*)d_in[1];    // [3072,1024]
  const float* Wout  = (const float*)d_in[2];    // [1024,1024]
  const int*   tpos  = (const int*)d_in[4];      // [2048]
  float*       out   = (float*)d_out;            // [4,2048,1024]

  char* ws = (char*)d_ws;
  const size_t OFF_XBF   = 0;                          // 16 MB
  const size_t OFF_WQKV  = OFF_XBF   + (size_t)ROWS * DM * 2;
  const size_t OFF_WOUT  = OFF_WQKV  + (size_t)QKVN * DM * 2;
  const size_t OFF_QKV   = OFF_WOUT  + (size_t)DM * DM * 2;
  const size_t OFF_Q     = OFF_QKV   + (size_t)ROWS * QKVN * 2;
  const size_t OFF_K     = OFF_Q     + (size_t)ROWS * DM * 2;
  const size_t OFF_V     = OFF_K     + (size_t)ROWS * DM * 2;
  const size_t OFF_ATTN  = OFF_V     + (size_t)ROWS * DM * 2;

  __bf16* xbf   = (__bf16*)(ws + OFF_XBF);
  __bf16* wqkvb = (__bf16*)(ws + OFF_WQKV);
  __bf16* woutb = (__bf16*)(ws + OFF_WOUT);
  __bf16* qkvb  = (__bf16*)(ws + OFF_QKV);
  __bf16* qb    = (__bf16*)(ws + OFF_Q);
  __bf16* kb    = (__bf16*)(ws + OFF_K);
  __bf16* vt    = (__bf16*)(ws + OFF_V);
  __bf16* attnb = (__bf16*)(ws + OFF_ATTN);

  // 1) converts
  {
    int n = ROWS * DM;
    f32_to_bf16_kernel<<<(n + 255) / 256, 256, 0, stream>>>(x, xbf, n);
    n = QKVN * DM;
    f32_to_bf16_kernel<<<(n + 255) / 256, 256, 0, stream>>>(Wqkv, wqkvb, n);
    n = DM * DM;
    f32_to_bf16_kernel<<<(n + 255) / 256, 256, 0, stream>>>(Wout, woutb, n);
  }

  // 2) qkv = x @ Wqkv^T  (8192 x 3072, K=1024)
  gemm_bt_kernel<__bf16><<<dim3(QKVN / 128, ROWS / 64), 256, 0, stream>>>(
      xbf, wqkvb, qkvb, ROWS, QKVN, DM);

  // 3) RoPE + head repack (+ V transpose)
  {
    int n = BATCH * SEQ * NH * 32;
    rope_pack_kernel<<<(n + 255) / 256, 256, 0, stream>>>(qkvb, tpos, qb, kb, vt);
  }

  // 4) causal flash attention
  attn_kernel<<<(BATCH * NH * 128) / 4, 128, 0, stream>>>(qb, kb, vt, attnb);

  // 5) out = attn @ Wout^T  (8192 x 1024, K=1024)
  gemm_bt_kernel<float><<<dim3(DM / 128, ROWS / 64), 256, 0, stream>>>(
      attnb, woutb, out, ROWS, DM, DM);
}
